// AttentionTIE_56435870270136
// MI455X (gfx1250) — matrix operand order repaired and checked by reading, not compiled, not run on hardware
//
#include <hip/hip_runtime.h>
#include <hip/hip_bf16.h>
#include <math.h>

#define TN 3072
#define TB 4
#define TC 128
#define EPSV 1e-5f
#define USE_ASYNC 1

typedef __attribute__((ext_vector_type(16))) __bf16 v16bf;
typedef __attribute__((ext_vector_type(8)))  float  v8f;

// Native conversion: lowers to hardware bf16 cvt (RNE) instead of manual
// bfe/add3/mov_b16 bit twiddling.
static __device__ __forceinline__ __bf16 f2bf(float f) { return (__bf16)f; }

union BF16x16 { v16bf v; __bf16 h[16]; uint4 q[2]; };

static __device__ __forceinline__ v8f wmma_bf16(v16bf a, v16bf b, v8f c) {
  // D = A(16x32 bf16) x B(32x16 bf16) + C(16x16 f32)
  return __builtin_amdgcn_wmma_f32_16x16x32_bf16(false, a, false, b, (short)0, c,
                                                 false, false);
}

static __device__ __forceinline__ v8f vzero8() {
  v8f z;
#pragma unroll
  for (int i = 0; i < 8; ++i) z[i] = 0.0f;
  return z;
}

// A-frag (16x32 bf16) from a bf16 row (contiguous K), contiguous k = 16*half + s order (same as B-frags):
// lane<16: K = kc*32 + {0..15}; lane>=16: K = kc*32 + {16..31}
static __device__ __forceinline__ v16bf load_afrag_bf16(const __bf16* rowp, int kc, int khi) {
  BF16x16 u;
  u.q[0] = *(const uint4*)(rowp + kc * 32 + 2 * khi);
  u.q[1] = *(const uint4*)(rowp + kc * 32 + 2 * khi + 8);
  return u.v;
}

// Same layout, sourced from f32 and converted to bf16.
static __device__ __forceinline__ v16bf load_afrag_f32(const float* rowp, int kc, int khi) {
  BF16x16 u;
#pragma unroll
  for (int e = 0; e < 8; ++e) {
    u.h[e]     = f2bf(rowp[kc * 32 + 2 * khi + e]);
    u.h[8 + e] = f2bf(rowp[kc * 32 + 2 * khi + 8 + e]);
  }
  return u.v;
}

// B-frag (32x16 bf16) for D = A @ W^T: lane holds weight-row n = (lane&15),
// K-half = lane>>4, 16 contiguous K elements from f32 weight row.
static __device__ __forceinline__ v16bf load_bfrag_w(const float* wrow, int kc, int khalf) {
  BF16x16 u;
#pragma unroll
  for (int i = 0; i < 16; ++i) u.h[i] = f2bf(wrow[kc * 32 + khalf * 16 + i]);
  return u.v;
}

// ---------------------------------------------------------------------------
// Kernel 1: v_r = x@recv_w^T + recv@mem_w^T + res_r ; v_s likewise ; q = x@q_w^T
// Also per-row stats: m_r, m_s, var_r, var_s, qsum, q.v_r
// One wave per 16-row tile (rows of one batch b).
// ---------------------------------------------------------------------------
__global__ void __launch_bounds__(256) kproj(
    const float* __restrict__ x, const float* __restrict__ recv,
    const float* __restrict__ send, const float* __restrict__ res_r,
    const float* __restrict__ res_s, const float* __restrict__ rel,
    const float* __restrict__ qw,
    float* __restrict__ vr_f, float* __restrict__ vs_f,
    __bf16* __restrict__ vr_b, __bf16* __restrict__ vs_b, __bf16* __restrict__ q_b,
    float* __restrict__ m_r, float* __restrict__ m_s,
    float* __restrict__ var_r, float* __restrict__ var_s,
    float* __restrict__ qsum, float* __restrict__ qdvr)
{
  const int lane  = threadIdx.x & 31;
  const int wave  = threadIdx.x >> 5;
  const int tile  = blockIdx.x * 8 + wave;
  const int tpb   = TN / 16;
  const int b     = tile / tpb;
  const int i0    = (tile % tpb) * 16;
  const int col   = lane & 15;
  const int khalf = lane >> 4;
  const int khi   = khalf * 8;

  // A-frags from (N,B,C)-layout inputs, row stride B*C
  const int arow = i0 + col;
  const float* xr = x    + (size_t)arow * TB * TC + b * TC;
  const float* rr = recv + (size_t)arow * TB * TC + b * TC;
  const float* sr = send + (size_t)arow * TB * TC + b * TC;

  v16bf xa[4], ra[4], sa[4];
#pragma unroll
  for (int kc = 0; kc < 4; ++kc) {
    xa[kc] = load_afrag_f32(xr, kc, khi);
    ra[kc] = load_afrag_f32(rr, kc, khi);
    sa[kc] = load_afrag_f32(sr, kc, khi);
  }

  float svr[8], svr2[8], svs[8], svs2[8], sq[8], sqv[8];
#pragma unroll
  for (int v = 0; v < 8; ++v) { svr[v]=svr2[v]=svs[v]=svs2[v]=sq[v]=sqv[v]=0.f; }

  const int rbase = i0 + khalf * 8;

  for (int dc = 0; dc < 8; ++dc) {
    const int n = dc * 16 + col;              // output feature row of weight
    const float* w_mem = rel + (size_t)n * (3 * TC);       // relation_attn[:, :C]
    const float* w_rcv = w_mem + TC;                        // [:, C:2C]
    const float* w_snd = w_mem + 2 * TC;                    // [:, 2C:]
    const float* w_q   = qw  + (size_t)n * TC;

    v8f aR = vzero8(), aS = vzero8(), aQ = vzero8();
#pragma unroll
    for (int kc = 0; kc < 4; ++kc) {
      v16bf wb;
      wb = load_bfrag_w(w_rcv, kc, khalf);  aR = wmma_bf16(xa[kc], wb, aR);
      wb = load_bfrag_w(w_mem, kc, khalf);  aR = wmma_bf16(ra[kc], wb, aR);
                                            aS = wmma_bf16(sa[kc], wb, aS);
      wb = load_bfrag_w(w_snd, kc, khalf);  aS = wmma_bf16(xa[kc], wb, aS);
      wb = load_bfrag_w(w_q,   kc, khalf);  aQ = wmma_bf16(xa[kc], wb, aQ);
    }

    const int d = dc * 16 + col;
#pragma unroll
    for (int v = 0; v < 8; ++v) {
      const int row = rbase + v;                     // D-tile row = v + 8*khalf
      const size_t gi = ((size_t)b * TN + row) * TC + d;
      const size_t ri = (size_t)row * TB * TC + b * TC + d;
      float vr = aR[v] + res_r[ri];
      float vs = aS[v] + res_s[ri];
      float qv = aQ[v];
      vr_f[gi] = vr;        vs_f[gi] = vs;
      vr_b[gi] = f2bf(vr);  vs_b[gi] = f2bf(vs);  q_b[gi] = f2bf(qv);
      svr[v]  += vr;  svr2[v] += vr * vr;
      svs[v]  += vs;  svs2[v] += vs * vs;
      sq[v]   += qv;  sqv[v]  += qv * vr;
    }
  }

#pragma unroll
  for (int m = 1; m <= 8; m <<= 1) {
#pragma unroll
    for (int v = 0; v < 8; ++v) {
      svr[v]  += __shfl_xor(svr[v],  m, 32);
      svr2[v] += __shfl_xor(svr2[v], m, 32);
      svs[v]  += __shfl_xor(svs[v],  m, 32);
      svs2[v] += __shfl_xor(svs2[v], m, 32);
      sq[v]   += __shfl_xor(sq[v],   m, 32);
      sqv[v]  += __shfl_xor(sqv[v],  m, 32);
    }
  }
  if (col == 0) {
#pragma unroll
    for (int v = 0; v < 8; ++v) {
      const int row = rbase + v;
      const size_t si = (size_t)b * TN + row;
      float mr = svr[v] / TC, ms = svs[v] / TC;
      m_r[si] = mr;  m_s[si] = ms;
      var_r[si] = svr2[v] / TC - mr * mr;
      var_s[si] = svs2[v] / TC - ms * ms;
      qsum[si] = sq[v];  qdvr[si] = sqv[v];
    }
  }
}

// ---------------------------------------------------------------------------
// Kernel 2: fused flash-style moment-normalized attention.
// One wave = 16 query rows; block (8 waves, same batch) cooperatively stages
// 32-row v_s tiles into LDS (row-major via ASYNC DMA + transposed) each step.
// ---------------------------------------------------------------------------
__global__ void __launch_bounds__(256) kattn(
    const __bf16* __restrict__ q_b, const __bf16* __restrict__ vr_b,
    const __bf16* __restrict__ vs_b, const float* __restrict__ vr_f,
    const float* __restrict__ m_r, const float* __restrict__ m_s,
    const float* __restrict__ var_r, const float* __restrict__ var_s,
    const float* __restrict__ qsum, const float* __restrict__ qdvr,
    const unsigned char* __restrict__ mask,
    const float* __restrict__ n_w, const float* __restrict__ n_b,
    __bf16* __restrict__ h_b)
{
  __shared__ __bf16 lds_rm[32 * 136];   // v_s tile, row-major (j, d), padded
  __shared__ __bf16 lds_tr[128 * 40];   // v_s tile, transposed (d, j), padded
  __shared__ __bf16 lds_p[8 * 16 * 40]; // per-wave staged P_std tile (i, j)
  __shared__ float lds_msv[32], lds_vsv[32];

  const int lane  = threadIdx.x & 31;
  const int wave  = threadIdx.x >> 5;
  const int tpb   = TN / 16;
  const int b     = (blockIdx.x * 8) / tpb;          // block-uniform batch
  const int tile  = blockIdx.x * 8 + wave;
  const int i0    = (tile % tpb) * 16;
  const int col   = lane & 15;
  const int khalf = lane >> 4;
  const int khi   = khalf * 8;
  const int rbase = i0 + khalf * 8;

  // q and v_r A-frags from bf16 workspace
  const __bf16* qrow = q_b  + ((size_t)b * TN + i0 + col) * TC;
  const __bf16* vrow = vr_b + ((size_t)b * TN + i0 + col) * TC;
  v16bf qa[4], va[4];
#pragma unroll
  for (int kc = 0; kc < 4; ++kc) {
    qa[kc] = load_afrag_bf16(qrow, kc, khi);
    va[kc] = load_afrag_bf16(vrow, kc, khi);
  }

  // per-row (this lane group's 8 rows) stats
  float mr8[8], vrv8[8], qs8[8], qv8[8], M8[8], L8[8];
#pragma unroll
  for (int v = 0; v < 8; ++v) {
    const size_t si = (size_t)b * TN + rbase + v;
    mr8[v] = m_r[si];  vrv8[v] = var_r[si];
    qs8[v] = qsum[si]; qv8[v]  = qdvr[si];
    M8[v] = -1e30f;    L8[v] = 0.0f;
  }
  v8f acc[9];                                  // 8 d-chunks + (ones, m_s) extra
#pragma unroll
  for (int a = 0; a < 9; ++a) acc[a] = vzero8();

  __bf16* pst = lds_p + wave * 16 * 40;
  const float invD2 = 2.0f / TC;
  const float scale = 0.08838834764831845f;    // 128^-0.5
  const unsigned char* mbase = mask + (size_t)b * TN * TN;

  for (int jb = 0; jb < TN; jb += 32) {
    __syncthreads();
    {  // cooperative staging of v_s[jb..jb+31, :] (bf16) into both layouts
      const int t = threadIdx.x;
      const int j = t >> 3;
      const int d0 = (t & 7) * 16;
      const __bf16* src = vs_b + ((size_t)b * TN + jb + j) * TC + d0;
#if USE_ASYNC
      // Async DMA global -> LDS (row-major copy), tracked by ASYNCcnt.
      {
        unsigned ldsoff = (unsigned)(size_t)(&lds_rm[j * 136 + d0]);
        asm volatile("global_load_async_to_lds_b128 %0, %1, off"
                     :: "v"(ldsoff), "v"(src) : "memory");
        asm volatile("global_load_async_to_lds_b128 %0, %1, off"
                     :: "v"(ldsoff + 16u), "v"(src + 8) : "memory");
      }
#endif
      uint4 p0 = *(const uint4*)(src);
      uint4 p1 = *(const uint4*)(src + 8);
#if !USE_ASYNC
      *(uint4*)(&lds_rm[j * 136 + d0])     = p0;
      *(uint4*)(&lds_rm[j * 136 + d0 + 8]) = p1;
#endif
      union { uint4 q; __bf16 h[8]; } a0, a1; a0.q = p0; a1.q = p1;
#pragma unroll
      for (int e = 0; e < 8; ++e) {
        lds_tr[(d0 + e) * 40 + j]     = a0.h[e];
        lds_tr[(d0 + 8 + e) * 40 + j] = a1.h[e];
      }
      if (t < 32) {
        lds_msv[t] = m_s[(size_t)b * TN + jb + t];
        lds_vsv[t] = var_s[(size_t)b * TN + jb + t];
      }
      if (jb + 32 < TN)   // hint next tile into cache (global_prefetch_b8)
        __builtin_prefetch(vs_b + ((size_t)b * TN + jb + 32 + j) * TC + d0, 0, 0);
    }
#if USE_ASYNC
    asm volatile("s_wait_asynccnt 0x0" ::: "memory");
#endif
    __syncthreads();

    // ---- S tiles: S1 = v_r . v_s^T, S2 = q . v_s^T for both 16-col halves
    v8f s1a = vzero8(), s2a = vzero8(), s1b = vzero8(), s2b = vzero8();
    {
      const __bf16* br0 = &lds_rm[(col)      * 136];
      const __bf16* br1 = &lds_rm[(16 + col) * 136];
#pragma unroll
      for (int kc = 0; kc < 4; ++kc) {
        BF16x16 u0, u1;
        u0.q[0] = *(const uint4*)(br0 + kc * 32 + khalf * 16);
        u0.q[1] = *(const uint4*)(br0 + kc * 32 + khalf * 16 + 8);
        u1.q[0] = *(const uint4*)(br1 + kc * 32 + khalf * 16);
        u1.q[1] = *(const uint4*)(br1 + kc * 32 + khalf * 16 + 8);
        s1a = wmma_bf16(va[kc], u0.v, s1a);
        s2a = wmma_bf16(qa[kc], u0.v, s2a);
        s1b = wmma_bf16(va[kc], u1.v, s1b);
        s2b = wmma_bf16(qa[kc], u1.v, s2b);
      }
    }

    // ---- element-wise: logits, std, mask
    const float ms0 = lds_msv[col],      vs0 = lds_vsv[col];
    const float ms1 = lds_msv[16 + col], vs1 = lds_vsv[16 + col];
    float lg0[8], lg1[8], is0[8], is1[8], mk0[8], mk1[8];
#pragma unroll
    for (int v = 0; v < 8; ++v) {
      const size_t mrow = (size_t)(rbase + v) * TN + jb;
      {
        float cross = invD2 * (s1a[v] - (float)TC * mr8[v] * ms0);
        float sd = sqrtf(vrv8[v] + vs0 + cross + EPSV);
        float lg = (qv8[v] + s2a[v] - qs8[v] * (mr8[v] + ms0)) / sd * scale;
        bool m = mbase[mrow + col] != 0;
        lg0[v] = m ? -1e30f : lg;  is0[v] = 1.0f / sd;  mk0[v] = m ? 0.f : 1.f;
      }
      {
        float cross = invD2 * (s1b[v] - (float)TC * mr8[v] * ms1);
        float sd = sqrtf(vrv8[v] + vs1 + cross + EPSV);
        float lg = (qv8[v] + s2b[v] - qs8[v] * (mr8[v] + ms1)) / sd * scale;
        bool m = mbase[mrow + 16 + col] != 0;
        lg1[v] = m ? -1e30f : lg;  is1[v] = 1.0f / sd;  mk1[v] = m ? 0.f : 1.f;
      }
    }

    // ---- online softmax: joint max over 32 cols, one rescale
#pragma unroll
    for (int v = 0; v < 8; ++v) {
      float tm = fmaxf(lg0[v], lg1[v]);
#pragma unroll
      for (int m = 1; m <= 8; m <<= 1) tm = fmaxf(tm, __shfl_xor(tm, m, 32));
      const float nM = fmaxf(M8[v], tm);
      const float al = __expf(M8[v] - nM);
      const float p0 = mk0[v] * __expf(lg0[v] - nM);
      const float p1 = mk1[v] * __expf(lg1[v] - nM);
      float sp = p0 + p1;
#pragma unroll
      for (int m = 1; m <= 8; m <<= 1) sp += __shfl_xor(sp, m, 32);
      L8[v] = L8[v] * al + sp;
      M8[v] = nM;
#pragma unroll
      for (int a = 0; a < 9; ++a) acc[a][v] *= al;
      // stage P_std = p/std (bf16) in D-layout for re-load as A-frag
      const int prow = v + 8 * khalf;
      pst[prow * 40 + col]      = f2bf(p0 * is0[v]);
      pst[prow * 40 + 16 + col] = f2bf(p1 * is1[v]);
    }

    // ---- reload P as A-frag, accumulate P_std @ [v_s | ones | m_s]
    v16bf pa;
    {
      BF16x16 u;
      const __bf16* pr = pst + col * 40;
      u.q[0] = *(const uint4*)(pr + 2 * khi);
      u.q[1] = *(const uint4*)(pr + 2 * khi + 8);
      pa = u.v;
    }
#pragma unroll
    for (int dc = 0; dc < 8; ++dc) {
      BF16x16 u;
      const __bf16* tr = &lds_tr[(dc * 16 + col) * 40 + khalf * 16];
      u.q[0] = *(const uint4*)(tr);
      u.q[1] = *(const uint4*)(tr + 8);
      acc[dc] = wmma_bf16(pa, u.v, acc[dc]);
    }
    {  // extra B: col0 = 1.0 -> sum(P_std); col1 = m_s_j -> sum(P_std*m_s)
      BF16x16 u;
#pragma unroll
      for (int i = 0; i < 16; ++i) {
        const int k = khalf * 16 + i;
        u.h[i] = (col == 0) ? (__bf16)1.0f
               : (col == 1) ? f2bf(lds_msv[k]) : (__bf16)0.0f;
      }
      acc[8] = wmma_bf16(pa, u.v, acc[8]);
    }
  }

  // ---- finalize ----
  float sps[8], spms[8], iL[8];
#pragma unroll
  for (int v = 0; v < 8; ++v) {
    sps[v]  = __shfl(acc[8][v], khalf * 16,     32);  // col 0
    spms[v] = __shfl(acc[8][v], khalf * 16 + 1, 32);  // col 1
    iL[v]   = 1.0f / L8[v];
  }
#pragma unroll
  for (int dc = 0; dc < 8; ++dc) {
    const int d = dc * 16 + col;
    const float nw = n_w[d], nb = n_b[d];
#pragma unroll
    for (int v = 0; v < 8; ++v) {
      const int row = rbase + v;
      const size_t gi = ((size_t)b * TN + row) * TC + d;
      const float vr = vr_f[gi];
      // out = (acc - Sum(pstd*mean) + Sum(pstd)*v_r) / L ;  mean = m_r + m_s
      float o = (acc[dc][v] - (mr8[v] * sps[v] + spms[v]) + sps[v] * vr) * iL[v];
      h_b[gi] = f2bf(o * nw + nb);
    }
  }
}

// ---------------------------------------------------------------------------
// Kernel 3: three output projections  out = in @ W^T + bias, written (N,B,C).
// blockIdx.y selects job: 0: h@proj_w, 1: v_r@r_w, 2: v_s@s_w.
// ---------------------------------------------------------------------------
__global__ void __launch_bounds__(256) kout(
    const __bf16* __restrict__ h_b, const __bf16* __restrict__ vr_b,
    const __bf16* __restrict__ vs_b,
    const float* __restrict__ pw, const float* __restrict__ pb,
    const float* __restrict__ rw, const float* __restrict__ rb,
    const float* __restrict__ sw, const float* __restrict__ sb,
    float* __restrict__ out)
{
  const int job = blockIdx.y;
  const __bf16* in; const float* W; const float* bias; float* o;
  if (job == 0)      { in = h_b;  W = pw; bias = pb; o = out; }
  else if (job == 1) { in = vr_b; W = rw; bias = rb; o = out + (size_t)TN * TB * TC; }
  else               { in = vs_b; W = sw; bias = sb; o = out + 2 * (size_t)TN * TB * TC; }

  const int lane  = threadIdx.x & 31;
  const int wave  = threadIdx.x >> 5;
  const int tile  = blockIdx.x * 8 + wave;
  const int tpb   = TN / 16;
  const int b     = tile / tpb;
  const int i0    = (tile % tpb) * 16;
  const int col   = lane & 15;
  const int khalf = lane >> 4;
  const int khi   = khalf * 8;
  const int rbase = i0 + khalf * 8;

  const __bf16* irow = in + ((size_t)b * TN + i0 + col) * TC;
  v16bf ia[4];
#pragma unroll
  for (int kc = 0; kc < 4; ++kc) ia[kc] = load_afrag_bf16(irow, kc, khi);

  for (int dc = 0; dc < 8; ++dc) {
    const int n = dc * 16 + col;
    const float* wrow = W + (size_t)n * TC;
    v8f a = vzero8();
#pragma unroll
    for (int kc = 0; kc < 4; ++kc)
      a = wmma_bf16(ia[kc], load_bfrag_w(wrow, kc, khalf), a);
    const int d = dc * 16 + col;
    const float bv = bias[d];
#pragma unroll
    for (int v = 0; v < 8; ++v) {
      const int row = rbase + v;
      o[(size_t)row * TB * TC + b * TC + d] = a[v] + bv;   // (N,B,C) layout
    }
  }
}

// ---------------------------------------------------------------------------
extern "C" void kernel_launch(void* const* d_in, const int* in_sizes, int n_in,
                              void* d_out, int out_size, void* d_ws, size_t ws_size,
                              hipStream_t stream) {
  (void)in_sizes; (void)n_in; (void)out_size; (void)ws_size;
  const float* x      = (const float*)d_in[0];
  const unsigned char* mask = (const unsigned char*)d_in[1];
  const float* recv   = (const float*)d_in[2];
  const float* send   = (const float*)d_in[3];
  const float* res_r  = (const float*)d_in[4];
  const float* res_s  = (const float*)d_in[5];
  const float* rel    = (const float*)d_in[6];
  const float* qw     = (const float*)d_in[7];
  const float* pw     = (const float*)d_in[8];
  const float* pb     = (const float*)d_in[9];
  const float* rw     = (const float*)d_in[10];
  const float* rbv    = (const float*)d_in[11];
  const float* sw     = (const float*)d_in[12];
  const float* sbv    = (const float*)d_in[13];
  const float* n_w    = (const float*)d_in[14];
  const float* n_b    = (const float*)d_in[15];
  float* out = (float*)d_out;

  const size_t nbc = (size_t)TB * TN * TC;
  char* w = (char*)d_ws;
  float* vr_f = (float*)w;  w += nbc * 4;
  float* vs_f = (float*)w;  w += nbc * 4;
  __bf16* vr_b = (__bf16*)w; w += nbc * 2;
  __bf16* vs_b = (__bf16*)w; w += nbc * 2;
  __bf16* q_b  = (__bf16*)w; w += nbc * 2;
  __bf16* h_b  = (__bf16*)w; w += nbc * 2;
  float* m_r   = (float*)w; w += (size_t)TB * TN * 4;
  float* m_s   = (float*)w; w += (size_t)TB * TN * 4;
  float* var_r = (float*)w; w += (size_t)TB * TN * 4;
  float* var_s = (float*)w; w += (size_t)TB * TN * 4;
  float* qsum  = (float*)w; w += (size_t)TB * TN * 4;
  float* qdvr  = (float*)w; w += (size_t)TB * TN * 4;

  const int nblocks = (TB * TN / 16) / 8;  // 96
  kproj<<<nblocks, 256, 0, stream>>>(x, recv, send, res_r, res_s, rel, qw,
                                     vr_f, vs_f, vr_b, vs_b, q_b,
                                     m_r, m_s, var_r, var_s, qsum, qdvr);
  kattn<<<nblocks, 256, 0, stream>>>(q_b, vr_b, vs_b, vr_f,
                                     m_r, m_s, var_r, var_s, qsum, qdvr,
                                     mask, n_w, n_b, h_b);
  kout<<<dim3(nblocks, 3), 256, 0, stream>>>(h_b, vr_b, vs_b,
                                             pw, pb, rw, rbv, sw, sbv, out);
}
